// DEDICOMDecoder_50895362458308
// MI455X (gfx1250) — matrix operand order repaired and checked by reading, not compiled, not run on hardware
//
#include <hip/hip_runtime.h>
#include <math.h>
#include <stdint.h>

// ---------------------------------------------------------------------------
// DEDICOM decoder for MI455X (gfx1250, wave32, WMMA + async global->LDS).
//   prep:  Rh/Rl = split(R), Wh/Wl = split(D*D) (W zero-padded to 1024 rows)
//   pass1: P = z_i ⊙ (z_j @ R^T), stored pre-split as Ph/Pl  [32768,512] bf16
//   pass2: scores = sigmoid(P @ W^T) -> d_out [32768,960] f32
// GEMMs use V_WMMA_F32_16X16X32_BF16 with bf16 hi/lo split (AhBh+AhBl+AlBh)
// for f32-class accuracy at bf16 matrix rate. Pre-split operands make the
// pass-2 staging a pure copy, done with GLOBAL_LOAD_ASYNC_TO_LDS_B128
// (ASYNCcnt) overlapped with the WMMAs of the in-flight K-step.
// ---------------------------------------------------------------------------

typedef __bf16 bf16;
typedef __attribute__((ext_vector_type(4)))  bf16  v4bf;
typedef __attribute__((ext_vector_type(8)))  bf16  v8bf;
typedef __attribute__((ext_vector_type(16))) bf16  v16bf;
typedef __attribute__((ext_vector_type(8)))  float v8f;

#define HIDDEN 512
#define NUM_SE 960
#define WPAD   1024   // W rows padded so async B copies never go OOB
#define BATCH  32768

constexpr int BM = 64;              // block rows (batch)
constexpr int BN = 128;             // block cols (8 waves x 16)
constexpr int BK = 32;              // K per step (one WMMA K)
constexpr int LSTR = BK + 8;        // 40 bf16 = 80B rows: conflict-free b128 reads
constexpr int THREADS = 256;        // 8 wave32 waves
constexpr int KSTEPS = HIDDEN / BK; // 16

__device__ __forceinline__ void split2(float x, bf16& h, bf16& l) {
  bf16 hh = (bf16)x;                // RTNE f32->bf16
  h = hh;
  l = (bf16)(x - (float)hh);        // residual: hi+lo ~16 mantissa bits
}

// Async 16B/lane global->LDS copy (ASYNCcnt). All lanes active at call sites.
__device__ __forceinline__ void async_b128(uint32_t lds_addr, uint64_t gaddr) {
  asm volatile("global_load_async_to_lds_b128 %0, %1, off"
               :: "v"(lds_addr), "v"(gaddr) : "memory");
}
__device__ __forceinline__ void wait_async() {
  asm volatile("s_wait_asynccnt 0" ::: "memory");
}

// ---------------- prep kernels: one-time operand splitting -----------------
__global__ void prep_R(const float* __restrict__ R,
                       bf16* __restrict__ Rh, bf16* __restrict__ Rl) {
  int idx = blockIdx.x * blockDim.x + threadIdx.x;    // 512*512
  bf16 h, l;
  split2(R[idx], h, l);
  Rh[idx] = h; Rl[idx] = l;
}
__global__ void prep_W(const float* __restrict__ D,
                       bf16* __restrict__ Wh, bf16* __restrict__ Wl) {
  int idx = blockIdx.x * blockDim.x + threadIdx.x;    // 1024*512
  int n = idx >> 9;
  float d = (n < NUM_SE) ? D[idx] : 0.f;
  bf16 h, l;
  split2(d * d, h, l);
  Wh[idx] = h; Wl[idx] = l;
}

// ---------------- GEMM kernel ----------------------------------------------
// MODE 0: A = z_j (f32, split in VALU), B = Rh/Rl; epilogue P=C*zi -> Ph/Pl
// MODE 1: A = Ph/Pl, B = Wh/Wl (pure async copies); epilogue sigmoid -> OutS
template <int MODE>
__global__ __launch_bounds__(THREADS)
void dedicom_gemm(const float* __restrict__ Af32,
                  const bf16* __restrict__ Aph, const bf16* __restrict__ Apl,
                  const bf16* __restrict__ Bh,  const bf16* __restrict__ Bl,
                  const float* __restrict__ Zi,
                  float* __restrict__ OutS,
                  bf16* __restrict__ OutPh, bf16* __restrict__ OutPl,
                  const int nRows, const int ldOut)
{
  __shared__ __align__(16) bf16 sAh[2][BM][LSTR];
  __shared__ __align__(16) bf16 sAl[2][BM][LSTR];
  __shared__ __align__(16) bf16 sBh[2][BN][LSTR];
  __shared__ __align__(16) bf16 sBl[2][BN][LSTR];

  const int t    = threadIdx.x;
  const int wave = t >> 5;            // 0..7 -> 16-column strip
  const int lane = t & 31;
  const int g    = (lane >> 4) & 1;   // half-wave group
  const int nl   = lane & 15;

  const int mBase = blockIdx.x * BM;
  const int nBase = blockIdx.y * BN;

  // ---- async copy geometry (16B chunks, constant over K steps) ----
  // B tile: 128 rows x 32 bf16 = 8KB per array -> 2 chunks/thread/array
  const int rowB0 = t >> 2,             colB0 = (t & 3) * 8;
  const int rowB1 = (t + THREADS) >> 2, colB1 = ((t + THREADS) & 3) * 8;
  // A tile (MODE 1): 64 rows x 32 bf16 = 4KB per array -> 1 chunk/thread
  const int rowA = t >> 2, colA = (t & 3) * 8;

  // LDS byte addresses (addrspace(3) offset = low 32 bits of generic ptr)
  const uint32_t aStride = BM * LSTR * (uint32_t)sizeof(bf16);   // 5120
  const uint32_t bStride = BN * LSTR * (uint32_t)sizeof(bf16);   // 10240
  const uint32_t bh0 = (uint32_t)(uintptr_t)&sBh[0][rowB0][colB0];
  const uint32_t bh1 = (uint32_t)(uintptr_t)&sBh[0][rowB1][colB1];
  const uint32_t bl0 = (uint32_t)(uintptr_t)&sBl[0][rowB0][colB0];
  const uint32_t bl1 = (uint32_t)(uintptr_t)&sBl[0][rowB1][colB1];
  const uint32_t ah0 = (uint32_t)(uintptr_t)&sAh[0][rowA][colA];
  const uint32_t al0 = (uint32_t)(uintptr_t)&sAl[0][rowA][colA];

  auto issueAsync = [&](int s, int buf) {
    const long k = (long)s * BK;
    const long ob0 = (long)(nBase + rowB0) * HIDDEN + k + colB0;
    const long ob1 = (long)(nBase + rowB1) * HIDDEN + k + colB1;
    async_b128(bh0 + buf * bStride, (uint64_t)(uintptr_t)(Bh + ob0));
    async_b128(bl0 + buf * bStride, (uint64_t)(uintptr_t)(Bl + ob0));
    async_b128(bh1 + buf * bStride, (uint64_t)(uintptr_t)(Bh + ob1));
    async_b128(bl1 + buf * bStride, (uint64_t)(uintptr_t)(Bl + ob1));
    if (MODE == 1) {
      const long oa = (long)(mBase + rowA) * HIDDEN + k + colA;
      async_b128(ah0 + buf * aStride, (uint64_t)(uintptr_t)(Aph + oa));
      async_b128(al0 + buf * aStride, (uint64_t)(uintptr_t)(Apl + oa));
    }
  };

  // ---- MODE 0: f32 A staging with VALU split (z_j must be converted) ----
  float4 aReg[2];                      // 64x32 f32 = 512 float4 / 256 thr
  auto loadA = [&](int s) {
#pragma unroll
    for (int i = 0; i < 2; ++i) {
      int idx = t + i * THREADS;
      int r = idx >> 3, c4 = idx & 7;
      aReg[i] = *(const float4*)(Af32 + (long)(mBase + r) * HIDDEN + s * BK + c4 * 4);
    }
  };
  auto stageA = [&](int buf) {
#pragma unroll
    for (int i = 0; i < 2; ++i) {
      int idx = t + i * THREADS;
      int r = idx >> 3, c = (idx & 7) * 4;
      float4 v = aReg[i];
      bf16 h0, l0, h1, l1, h2, l2, h3, l3;
      split2(v.x, h0, l0); split2(v.y, h1, l1);
      split2(v.z, h2, l2); split2(v.w, h3, l3);
      *(v4bf*)&sAh[buf][r][c] = (v4bf){h0, h1, h2, h3};
      *(v4bf*)&sAl[buf][r][c] = (v4bf){l0, l1, l2, l3};
    }
  };

  // A operand per §7.12.2 (16-bit A 16x32): lane row = L%16,
  // K = {8g..8g+7} in v[0:3] and {16+8g..16+8g+7} in v[4:7].
  auto mkA = [&](const bf16* rowp) -> v16bf {
    v8bf lo = *(const v8bf*)(rowp + g * 8);
    v8bf hi = *(const v8bf*)(rowp + 16 + g * 8);
    return __builtin_shufflevector(lo, hi, 0, 1, 2, 3, 4, 5, 6, 7,
                                   8, 9, 10, 11, 12, 13, 14, 15);
  };

  v8f acc[4];
  const v8f vzero = {0.f, 0.f, 0.f, 0.f, 0.f, 0.f, 0.f, 0.f};
#pragma unroll
  for (int i = 0; i < 4; ++i) acc[i] = vzero;

  auto compute = [&](int buf) {
    // B operand (16-bit B 32x16): lane col = L%16 holds
    // K = {8g..8g+7} in v[0:3] and {16+8g..16+8g+7} in v[4:7] (same as A).
    const int bcol = wave * 16 + nl;
    v16bf bh = mkA(&sBh[buf][bcol][0]);
    v16bf bl = mkA(&sBl[buf][bcol][0]);
#pragma unroll
    for (int mt = 0; mt < 4; ++mt) {
      v16bf ah = mkA(&sAh[buf][mt * 16 + nl][0]);
      v16bf al = mkA(&sAl[buf][mt * 16 + nl][0]);
      acc[mt] = __builtin_amdgcn_wmma_f32_16x16x32_bf16(
          false, ah, false, bh, (short)0, acc[mt], false, false);
      acc[mt] = __builtin_amdgcn_wmma_f32_16x16x32_bf16(
          false, ah, false, bl, (short)0, acc[mt], false, false);
      acc[mt] = __builtin_amdgcn_wmma_f32_16x16x32_bf16(
          false, al, false, bh, (short)0, acc[mt], false, false);
    }
  };

  // ---- software pipeline: async copies for step s+1 overlap WMMAs of s ----
  issueAsync(0, 0);
  if (MODE == 0) { loadA(0); stageA(0); }
  wait_async();
  __syncthreads();
  for (int s = 0; s < KSTEPS; ++s) {
    const int cur = s & 1;
    if (s + 1 < KSTEPS) {
      issueAsync(s + 1, cur ^ 1);
      if (MODE == 0) loadA(s + 1);
    }
    compute(cur);
    if (MODE == 0 && s + 1 < KSTEPS) stageA(cur ^ 1);
    wait_async();
    __syncthreads();
  }

  // C/D layout: lane L, VGPR r -> (M = r + 8*(L/16), N = L%16).
  const int nOut = nBase + wave * 16 + nl;
#pragma unroll
  for (int mt = 0; mt < 4; ++mt) {
#pragma unroll
    for (int r = 0; r < 8; ++r) {
      const long m = mBase + mt * 16 + g * 8 + r;
      float v = acc[mt][r];
      if (MODE == 0) {
        float p = v * Zi[m * (long)HIDDEN + nOut];
        bf16 h, l;
        split2(p, h, l);
        OutPh[m * (long)HIDDEN + nOut] = h;
        OutPl[m * (long)HIDDEN + nOut] = l;
      } else {
        if (nOut < nRows)
          OutS[m * (long)ldOut + nOut] = 1.0f / (1.0f + __expf(-v));
      }
    }
  }
}

extern "C" void kernel_launch(void* const* d_in, const int* in_sizes, int n_in,
                              void* d_out, int out_size, void* d_ws, size_t ws_size,
                              hipStream_t stream) {
  const float* z_i = (const float*)d_in[0];
  const float* z_j = (const float*)d_in[1];
  const float* R   = (const float*)d_in[2];
  const float* D   = (const float*)d_in[3];
  float* out = (float*)d_out;

  // workspace carve-up (~67 MB; L2-resident on MI455X's 192 MB L2)
  char* ws = (char*)d_ws;
  bf16* Ph = (bf16*)ws;                                    // 32 MB
  bf16* Pl = (bf16*)(ws + (size_t)BATCH * HIDDEN * 2);     // 32 MB
  bf16* Rh = (bf16*)(ws + (size_t)BATCH * HIDDEN * 4);
  bf16* Rl = Rh + (size_t)HIDDEN * HIDDEN;
  bf16* Wh = Rl + (size_t)HIDDEN * HIDDEN;
  bf16* Wl = Wh + (size_t)WPAD * HIDDEN;

  prep_R<<<(HIDDEN * HIDDEN) / THREADS, THREADS, 0, stream>>>(R, Rh, Rl);
  prep_W<<<(WPAD * HIDDEN) / THREADS, THREADS, 0, stream>>>(D, Wh, Wl);

  dim3 blk(THREADS);
  dim3 g1(BATCH / BM, HIDDEN / BN);   // 512 x 4
  dedicom_gemm<0><<<g1, blk, 0, stream>>>(z_j, nullptr, nullptr, Rh, Rl,
                                          z_i, nullptr, Ph, Pl, HIDDEN, HIDDEN);
  dim3 g2(BATCH / BM, WPAD / BN);     // 512 x 8 (rows 960..1023 are zero-padded)
  dedicom_gemm<1><<<g2, blk, 0, stream>>>(nullptr, Ph, Pl, Wh, Wl,
                                          nullptr, out, nullptr, nullptr,
                                          NUM_SE, NUM_SE);
}
